// ActorCriticSpeakerDenseQuantized_47356309405909
// MI455X (gfx1250) — compile-verified, matrix operand order checked by reading, not run on hardware
//
#include <hip/hip_runtime.h>
#include <stdint.h>

typedef __attribute__((ext_vector_type(16))) _Float16 v16h;
typedef __attribute__((ext_vector_type(8)))  _Float16 v8h;
typedef __attribute__((ext_vector_type(4)))  _Float16 v4h;
typedef __attribute__((ext_vector_type(8)))  float    v8f;

#define NBATCH 131072
#define TILES_PER_BLOCK 4
#define NBLOCKS (NBATCH / (128 * TILES_PER_BLOCK))   // 256 blocks x 8 waves

// ---------------- LDS layout (units: _Float16 elements) ----------------
// weight blocks PRE-SWIZZLED into WMMA B-fragment order:
// element index = ((kc*NT + nt)*32 + lane)*16 + i   (32B contiguous per lane)
#define oW1   0          // 128x128
#define oW2   16384      // 128x128
#define oW3   32768      // 128x128
#define oWp   49152      // 128x64
#define oCB   57344      // 512x64 codebook (as B: K=64, N=512)
#define oWa   90112      // 64x128
#define oWs   98304      // 64x128
#define oWc1  106496     // 64x128
#define oWc3  114688     // 128x32
#define H_W_TOTAL 118784 // swizzled-weight halfs (one TDM load ws -> LDS)
#define oBufA 118784     // 8 waves x (16x128) activations, row-major
#define oBufB 135168     // 8 waves x (16x128)
#define oQ    151552     // 8 waves x (16x64)
#define H_TOTAL 159744   // halfs -> 319488 bytes

// float section (units: float), starts at byte H_TOTAL*2
#define fB1   0
#define fB2   128
#define fB3   256
#define fBp   384
#define fBa   448
#define fBs   576
#define fBc1  704
#define fBc2  832
#define fBc3  960
#define fBc4  992
#define fWc4  993        // 32 floats
#define fCBn  1025       // 512 floats: ||codebook_j||^2
#define F_TOTAL 1537

#define I_OFF_BYTES (H_TOTAL*2 + F_TOTAL*4)
#define SMEM_BYTES  (I_OFF_BYTES + 8*16*4)   // 326148 B <= 320KB WGP LDS

// ---------------- workspace layout (d_ws) ----------------
#define WS_NORM_OFF_BYTES (H_W_TOTAL * 2)    // 512 f32 codebook norms after halfs

// ---------------- output layout ----------------
#define AM_OFF  ((size_t)0)
#define SD_OFF  ((size_t)NBATCH*128)
#define CR_OFF  ((size_t)2*NBATCH*128)
#define LOSS_OFF (CR_OFF + NBATCH)
#define IDX_OFF  (LOSS_OFF + 1)

// ---- B-fragment swizzle: where element (k,n) of a KxN weight goes ----
__device__ __forceinline__ int swz_idx(int k, int n, int NT) {
  const int kc = k >> 5, kr = k & 31;
  const int i  = ((kr >> 4) << 3) | (kr & 7);
  const int L  = (((kr >> 3) & 1) << 4) | (n & 15);
  const int nt = n >> 4;
  return (((kc * NT + nt) * 32) + L) * 16 + i;
}

__device__ __forceinline__ void prep_w(_Float16* dst, const float* src,
                                       int K, int lgN, int gtid, int gstride) {
  const int N = 1 << lgN, NT = N >> 4;
  for (int s = gtid; s < (K << lgN); s += gstride)
    dst[swz_idx(s >> lgN, s & (N - 1), NT)] = (_Float16)src[s];
}

// one-time: swizzle all weights f32 -> f16 into d_ws (amortized over whole grid)
__global__ void acq_prep_kernel(const float* W1, const float* W2, const float* W3,
                                const float* Wp, const float* cb, const float* Wa,
                                const float* Ws, const float* Wc1, const float* Wc3,
                                _Float16* wsH, float* wsNorm, float* out) {
  const int gtid = blockIdx.x * 256 + threadIdx.x, gstride = gridDim.x * 256;
  prep_w(wsH + oW1,  W1,  128, 7, gtid, gstride);
  prep_w(wsH + oW2,  W2,  128, 7, gtid, gstride);
  prep_w(wsH + oW3,  W3,  128, 7, gtid, gstride);
  prep_w(wsH + oWp,  Wp,  128, 6, gtid, gstride);
  prep_w(wsH + oWa,  Wa,   64, 7, gtid, gstride);
  prep_w(wsH + oWs,  Ws,   64, 7, gtid, gstride);
  prep_w(wsH + oWc1, Wc1,  64, 7, gtid, gstride);
  prep_w(wsH + oWc3, Wc3, 128, 5, gtid, gstride);
  for (int s = gtid; s < 512 * 64; s += gstride)           // codebook stored [n][k]
    wsH[oCB + swz_idx(s & 63, s >> 6, 32)] = (_Float16)cb[s];
  for (int j = gtid; j < 512; j += gstride) {              // ||codebook_j||^2
    float s = 0.f;
#pragma unroll 8
    for (int k = 0; k < 64; ++k) { float v = cb[j * 64 + k]; s += v * v; }
    wsNorm[j] = s;
  }
  if (gtid == 0) out[LOSS_OFF] = 0.f;
}

// ---- TDM: one tensor_load_to_lds moves the whole 232KB weight blob ----
// D# per CDNA5 ISA 8.3/8.4: group0 = {flags, lds_addr, global_addr, type=2},
// group1 = {data_size=8B, tensor_dim0=tile_dim0=29696, dim1=1, stride0}.
__device__ __forceinline__ void tdm_stage_weights(char* ldsDst, const _Float16* wsH) {
  typedef __attribute__((ext_vector_type(4))) unsigned int u32x4;
  typedef __attribute__((ext_vector_type(8))) int i32x8;
  typedef __attribute__((ext_vector_type(4))) int i32x4;
  const unsigned long long ga = (unsigned long long)(uintptr_t)wsH;
  const unsigned lds = (unsigned)(uintptr_t)ldsDst;       // LDS aperture: addr[31:0]
  const unsigned n8 = (unsigned)(H_W_TOTAL * 2 / 8);      // 29696 x 8-byte elements
  u32x4 g0;
  g0[0] = 1u;                                             // count=1 (valid user D#)
  g0[1] = lds;                                            // lds_addr (bytes)
  g0[2] = (unsigned)(ga & 0xffffffffu);                   // global_addr[31:0]
  g0[3] = (unsigned)((ga >> 32) & 0x01ffffffu) | 0x80000000u; // addr[56:32] | type=2
  i32x8 g1;
  g1[0] = (int)(3u << 16);                                // data_size=3 (8B), no mcast
  g1[1] = (int)((n8 & 0xffffu) << 16);                    // tensor_dim0[15:0]
  g1[2] = (int)(((n8 >> 16) & 0xffffu) | (1u << 16));     // dim0[31:16] | tensor_dim1=1
  g1[3] = (int)((n8 & 0xffffu) << 16);                    // tile_dim0 = n8
  g1[4] = 1;                                              // tile_dim1 = 1
  g1[5] = (int)n8;                                        // tensor_dim0_stride[31:0]
  g1[6] = 0; g1[7] = 0;
  i32x4 z4; z4[0] = z4[1] = z4[2] = z4[3] = 0;            // groups 2/3 unused (<=2D)
#if __clang_major__ >= 23
  i32x8 z8; for (int i = 0; i < 8; ++i) z8[i] = 0;
  __builtin_amdgcn_tensor_load_to_lds(g0, g1, z4, z4, z8, 0);
#else
  __builtin_amdgcn_tensor_load_to_lds(g0, g1, z4, z4, 0);
#endif
}

// A fragment (16x32 f16) from row-major 16xldA LDS: two ds_load_b128 per lane
__device__ __forceinline__ v16h load_a_frag(const _Float16* base, int ldA, int kc, int lane) {
  const int m = lane & 15, hh = lane >> 4;
  const _Float16* p = base + m * ldA + kc * 32 + hh * 8;
  const v8h lo = *(const v8h*)(p);
  const v8h hi = *(const v8h*)(p + 16);
  v16h a;
#pragma unroll
  for (int i = 0; i < 8; ++i) { a[i] = lo[i]; a[i + 8] = hi[i]; }
  return a;
}

// B fragment from pre-swizzled LDS: 32 contiguous bytes per lane
__device__ __forceinline__ v16h load_b_swz(const _Float16* w, int NT, int kc, int nt, int lane) {
  return *(const v16h*)(w + (((kc * NT + nt) * 32) + lane) * 16);
}

// B fragment straight from global f32 [K][N] (Wc2 stays in L2 to fit LDS)
__device__ __forceinline__ v16h load_b_glob(const float* w, int N, int kc, int nt, int lane) {
  const int n = nt * 16 + (lane & 15), hh = lane >> 4;
  v16h b;
#pragma unroll
  for (int i = 0; i < 16; ++i) {
    const int k = kc * 32 + ((i >> 3) << 4) + hh * 8 + (i & 7);
    b[i] = (_Float16)w[k * N + n];
  }
  return b;
}

// One wave computes a 16xN tile: D = act(A[16xK] * W[KxN] + bias)
// A-fragments hoisted to registers, reused across all N-tiles.
// ACT: 0=relu 1=none 2=sigmoid 3=sigmoid+1e-8
template<int K, int N, int ACT, bool LDSW>
__device__ __forceinline__ void dense16(const _Float16* aSrc, const _Float16* wL, const float* wG,
                                        const float* bias, _Float16* dst,
                                        float* gOut, size_t gBase, int r0, int lane)
{
  const int hh = lane >> 4, nn = lane & 15;
  v16h afrag[K / 32];
#pragma unroll
  for (int kc = 0; kc < K / 32; ++kc) afrag[kc] = load_a_frag(aSrc, K, kc, lane);
  for (int nt = 0; nt < N / 16; ++nt) {
    v8f c;
#pragma unroll
    for (int v = 0; v < 8; ++v) c[v] = 0.f;        // SRC2 becomes inline 0
#pragma unroll
    for (int kc = 0; kc < K / 32; ++kc) {
      v16h b;
      if constexpr (LDSW) b = load_b_swz(wL, N / 16, kc, nt, lane);
      else                b = load_b_glob(wG, N, kc, nt, lane);
      c = __builtin_amdgcn_wmma_f32_16x16x32_f16(false, afrag[kc], false, b, (short)0, c, false, false);
    }
    const float bv = bias[nt * 16 + nn];
#pragma unroll
    for (int v = 0; v < 8; ++v) {
      float x = c[v] + bv;
      if (ACT == 0)      x = x > 0.f ? x : 0.f;
      else if (ACT == 2) x = 1.f / (1.f + __expf(-x));
      else if (ACT == 3) x = 1.f / (1.f + __expf(-x)) + 1e-8f;
      const int m = v + hh * 8;                    // C/D layout: VGPR v -> row
      if (dst)  dst[m * N + nt * 16 + nn] = (_Float16)x;
      if (gOut) gOut[gBase + (size_t)(r0 + m) * N + nt * 16 + nn] = x;
    }
  }
}

__global__ void __launch_bounds__(256)
acq_fused_kernel(const int* __restrict__ obs, const float* __restrict__ embed,
                 const float* b1, const float* b2, const float* b3, const float* bp,
                 const float* __restrict__ cb, const float* ba, const float* bs,
                 const float* bc1, const float* Wc2, const float* bc2,
                 const float* bc3, const float* Wc4, const float* bc4,
                 const _Float16* __restrict__ wsH, const float* __restrict__ wsNorm,
                 float* __restrict__ out)
{
  extern __shared__ char smem[];
  _Float16* sh  = (_Float16*)smem;
  float*    sf  = (float*)(smem + H_TOTAL * 2);
  int*      sIx = (int*)(smem + I_OFF_BYTES);
  const int tid = threadIdx.x;

  // ---- stage pre-swizzled weights with one Tensor-DMA load (wave 0 issues) ----
  if (tid < 32) {
    tdm_stage_weights((char*)smem, wsH);
    __builtin_amdgcn_s_wait_tensorcnt(0);
  }
  for (int i = tid; i < 128; i += 256) { sf[fB1+i]=b1[i]; sf[fB2+i]=b2[i]; sf[fB3+i]=b3[i];
                                         sf[fBa+i]=ba[i]; sf[fBs+i]=bs[i];
                                         sf[fBc1+i]=bc1[i]; sf[fBc2+i]=bc2[i]; }
  for (int i = tid; i < 64;  i += 256) sf[fBp+i]  = bp[i];
  for (int i = tid; i < 32;  i += 256) { sf[fBc3+i] = bc3[i]; sf[fWc4+i] = Wc4[i]; }
  for (int j = tid; j < 512; j += 256) sf[fCBn + j] = wsNorm[j];
  if (tid == 0) sf[fBc4] = bc4[0];
  __syncthreads();

  const int wave = tid >> 5, lane = tid & 31;
  const int hh = lane >> 4, nn = lane & 15;
  _Float16* bufA = sh + oBufA + wave * 2048;   // 16x128
  _Float16* bufB = sh + oBufB + wave * 2048;   // 16x128
  _Float16* bufQ = sh + oQ    + wave * 1024;   // 16x64
  int* myIdx = sIx + wave * 16;

  for (int tile = 0; tile < TILES_PER_BLOCK; ++tile) {
    const int r0 = (blockIdx.x * TILES_PER_BLOCK + tile) * 128 + wave * 16;

    // ---- embedding gather: float4 loads -> packed f16 ds_store_b64 ----
    if (lane < 16) myIdx[lane] = obs[r0 + lane];
    __syncthreads();
    for (int c = lane; c < 16 * 32; c += 32) {
      const int m = c >> 5, k4 = (c & 31) << 2;
      const float4 f = *(const float4*)(embed + (size_t)myIdx[m] * 128 + k4);
      v4h h; h[0] = (_Float16)f.x; h[1] = (_Float16)f.y; h[2] = (_Float16)f.z; h[3] = (_Float16)f.w;
      *(v4h*)(bufA + m * 128 + k4) = h;
    }
    __syncthreads();

    // ---- dense stack (all WMMA) ----
    dense16<128,128,0,true>(bufA, sh+oW1, nullptr, sf+fB1, bufB, nullptr, 0, r0, lane); __syncthreads();
    dense16<128,128,0,true>(bufB, sh+oW2, nullptr, sf+fB2, bufA, nullptr, 0, r0, lane); __syncthreads();
    dense16<128,128,0,true>(bufA, sh+oW3, nullptr, sf+fB3, bufB, nullptr, 0, r0, lane); __syncthreads();
    dense16<128, 64,1,true>(bufB, sh+oWp, nullptr, sf+fBp, bufQ, nullptr, 0, r0, lane); __syncthreads();

    // ---- VQ: argmin_j ( ||cb_j||^2 - 2 z.cb_j ) via WMMA z @ cb^T ----
    float bestV[8]; int bestI[8];
#pragma unroll
    for (int v = 0; v < 8; ++v) { bestV[v] = 3.4e38f; bestI[v] = 0; }
    {
      v16h a0 = load_a_frag(bufQ, 64, 0, lane);
      v16h a1 = load_a_frag(bufQ, 64, 1, lane);
#pragma unroll 4
      for (int nt = 0; nt < 32; ++nt) {
        v8f c;
#pragma unroll
        for (int v = 0; v < 8; ++v) c[v] = 0.f;
        c = __builtin_amdgcn_wmma_f32_16x16x32_f16(false, a0, false, load_b_swz(sh+oCB, 32, 0, nt, lane), (short)0, c, false, false);
        c = __builtin_amdgcn_wmma_f32_16x16x32_f16(false, a1, false, load_b_swz(sh+oCB, 32, 1, nt, lane), (short)0, c, false, false);
        const int n = nt * 16 + nn;
        const float cn = sf[fCBn + n];
#pragma unroll
        for (int v = 0; v < 8; ++v) {
          const float score = cn - 2.f * c[v];
          if (score < bestV[v]) { bestV[v] = score; bestI[v] = n; }
        }
      }
    }
#pragma unroll
    for (int off = 8; off >= 1; off >>= 1) {        // reduce over 16 lanes of each half
#pragma unroll
      for (int v = 0; v < 8; ++v) {
        const float ov = __shfl_xor(bestV[v], off, 32);
        const int   oi = __shfl_xor(bestI[v], off, 32);
        if (ov < bestV[v] || (ov == bestV[v] && oi < bestI[v])) { bestV[v] = ov; bestI[v] = oi; }
      }
    }
    __syncthreads();
    if (nn == 0) {
#pragma unroll
      for (int v = 0; v < 8; ++v) {
        const int m = v + hh * 8;
        myIdx[m] = bestI[v];
        out[IDX_OFF + r0 + m] = (float)bestI[v];
      }
    }
    __syncthreads();

    // ---- quantize: lane -> fixed (row m, 32-wide k slab); float4 gathers ----
    {
      const int m = lane >> 1, kb = (lane & 1) << 5;
      const int ci = myIdx[m];
      float esum = 0.f;
#pragma unroll
      for (int t = 0; t < 8; ++t) {
        const float4 q4 = *(const float4*)(cb + (size_t)ci * 64 + kb + t * 4);
        const v4h z4 = *(const v4h*)(bufQ + m * 64 + kb + t * 4);
        float d0 = q4.x - (float)z4[0], d1 = q4.y - (float)z4[1];
        float d2 = q4.z - (float)z4[2], d3 = q4.w - (float)z4[3];
        esum += d0*d0 + d1*d1 + d2*d2 + d3*d3;
        v4h h; h[0]=(_Float16)q4.x; h[1]=(_Float16)q4.y; h[2]=(_Float16)q4.z; h[3]=(_Float16)q4.w;
        *(v4h*)(bufQ + m * 64 + kb + t * 4) = h;   // bufQ now holds q
      }
#pragma unroll
      for (int off = 16; off >= 1; off >>= 1) esum += __shfl_xor(esum, off, 32);
      if (lane == 0) atomicAdd(&out[LOSS_OFF], esum * (1.25f / ((float)NBATCH * 64.f)));
    }
    __syncthreads();

    // ---- heads ----
    dense16<64,128,2,true>(bufQ, sh+oWa,  nullptr, sf+fBa,  nullptr, out, AM_OFF, r0, lane);
    dense16<64,128,3,true>(bufQ, sh+oWs,  nullptr, sf+fBs,  nullptr, out, SD_OFF, r0, lane);
    dense16<64,128,2,true>(bufQ, sh+oWc1, nullptr, sf+fBc1, bufA, nullptr, 0, r0, lane); __syncthreads();
    dense16<128,128,2,false>(bufA, nullptr, Wc2,   sf+fBc2, bufB, nullptr, 0, r0, lane); __syncthreads();
    dense16<128,32,2,true>(bufB, sh+oWc3, nullptr, sf+fBc3, bufA, nullptr, 0, r0, lane); __syncthreads();

    if (lane < 16) {                                 // final 32 -> 1
      float s = sf[fBc4];
#pragma unroll
      for (int k = 0; k < 32; ++k) s += (float)bufA[lane * 32 + k] * sf[fWc4 + k];
      out[CR_OFF + r0 + lane] = s;
    }
    __syncthreads();                                 // tile buffers reused next iter
  }
}

extern "C" void kernel_launch(void* const* d_in, const int* in_sizes, int n_in,
                              void* d_out, int out_size, void* d_ws, size_t ws_size,
                              hipStream_t stream) {
  (void)in_sizes; (void)n_in; (void)out_size; (void)ws_size;
  const int*   obs  = (const int*)  d_in[0];
  const float* embed= (const float*)d_in[1];
  const float* W1 = (const float*)d_in[2];  const float* b1 = (const float*)d_in[3];
  const float* W2 = (const float*)d_in[4];  const float* b2 = (const float*)d_in[5];
  const float* W3 = (const float*)d_in[6];  const float* b3 = (const float*)d_in[7];
  const float* Wp = (const float*)d_in[8];  const float* bp = (const float*)d_in[9];
  const float* cb = (const float*)d_in[10];
  const float* Wa = (const float*)d_in[11]; const float* ba = (const float*)d_in[12];
  const float* Ws = (const float*)d_in[13]; const float* bs = (const float*)d_in[14];
  const float* Wc1= (const float*)d_in[15]; const float* bc1= (const float*)d_in[16];
  const float* Wc2= (const float*)d_in[17]; const float* bc2= (const float*)d_in[18];
  const float* Wc3= (const float*)d_in[19]; const float* bc3= (const float*)d_in[20];
  const float* Wc4= (const float*)d_in[21]; const float* bc4= (const float*)d_in[22];
  float* out = (float*)d_out;
  _Float16* wsH   = (_Float16*)d_ws;
  float*    wsNorm= (float*)((char*)d_ws + WS_NORM_OFF_BYTES);

  acq_prep_kernel<<<dim3(64), dim3(256), 0, stream>>>(W1, W2, W3, Wp, cb, Wa, Ws, Wc1, Wc3,
                                                      wsH, wsNorm, out);
  acq_fused_kernel<<<dim3(NBLOCKS), dim3(256), SMEM_BYTES, stream>>>(
      obs, embed, b1, b2, b3, bp, cb, ba, bs, bc1, Wc2, bc2, bc3, Wc4, bc4,
      wsH, wsNorm, out);
}
